// Net_37529424232709
// MI455X (gfx1250) — matrix operand order repaired and checked by reading, not compiled, not run on hardware
//
#include <hip/hip_runtime.h>

typedef __attribute__((ext_vector_type(16))) _Float16 v16h;
typedef __attribute__((ext_vector_type(8)))  float    v8f;

#define NEG_SLOPE 0.2f
#define MAXDEG 64
#define MAXH 2

// ---------------------------------------------------------------------------
// Weight prep: Wt[c][k] = f16(W[k][c]) for c < C, else 0.  C padded to C32.
// Transposed so each WMMA lane's 16 B elements are contiguous in memory.
// ---------------------------------------------------------------------------
__global__ void convert_weights(const float* __restrict__ W,
                                _Float16* __restrict__ Wt,
                                int K, int C, int C32) {
    const int idx = blockIdx.x * blockDim.x + threadIdx.x;
    if (idx >= C32 * K) return;
    const int c = idx / K;
    const int k = idx % K;
    Wt[idx] = (c < C) ? (_Float16)W[(size_t)k * C + c] : (_Float16)0.0f;
}

// ---------------------------------------------------------------------------
// GEMM: Y[N,C] = X[N,K] @ W[K,C] using f16 WMMA 16x16x32, fp32 accumulate.
// One wave computes a 16x32 output tile (two 16x16 WMMA tiles sharing the
// A fragment).  K must be a multiple of 32 (here K=128).
// ---------------------------------------------------------------------------
__global__ void gemm_wmma_f16(const float* __restrict__ X,
                              const _Float16* __restrict__ Wt,  // [C32][K], zero-padded
                              float* __restrict__ Y,
                              int N, int K, int C, int C32) {
    const int lane   = threadIdx.x & 31;
    const int wib    = threadIdx.x >> 5;
    const int tilesP = C32 >> 5;                  // 32-wide column pairs
    const int tilesM = (N + 15) >> 4;
    const long tile  = (long)blockIdx.x * (blockDim.x >> 5) + wib;
    if (tile >= (long)tilesM * tilesP) return;    // wave-uniform exit

    const int tm = (int)(tile / tilesP);
    const int tp = (int)(tile % tilesP);
    const int R  = tm << 4;
    const int Cb = tp << 5;

    // A fragment (16-bit A 16x32): lane<16 -> M=lane, K-base 0; lane>=16 -> K-base 16
    int arow = R + (lane & 15);
    if (arow >= N) arow = N - 1;                  // safe read; store is guarded
    const int akb = (lane >> 4) << 4;             // 0 or 16
    // B fragment (16-bit B 32x16): lane&15 = N col, lane>=16 -> K-base 16
    const int bkb = (lane >> 4) << 4;             // 0 or 16
    const float*    aprow = X  + (size_t)arow * K + akb;
    const _Float16* bp0   = Wt + (size_t)(Cb + (lane & 15)) * K + bkb;
    const _Float16* bp1   = bp0 + (size_t)16 * K;

    v8f acc0 = {}, acc1 = {};
    for (int k0 = 0; k0 < K; k0 += 32) {
        v16h a, b0, b1;
        const float* ap = aprow + k0;
#pragma unroll
        for (int e = 0; e < 8; ++e) {
            a[e]     = (_Float16)ap[e];           // K = k0+akb+e
            a[e + 8] = (_Float16)ap[e + 8];       // K = k0+akb+8+e
        }
#pragma unroll
        for (int e = 0; e < 16; ++e) {            // contiguous: 2x b128 each
            b0[e] = bp0[k0 + e];
            b1[e] = bp1[k0 + e];
        }
        acc0 = __builtin_amdgcn_wmma_f32_16x16x32_f16(
                   false, a, false, b0, (short)0, acc0, false, false);
        acc1 = __builtin_amdgcn_wmma_f32_16x16x32_f16(
                   false, a, false, b1, (short)0, acc1, false, false);
    }

    // D layout: VGPR r -> M = r + 8*(lane>=16), N = lane&15
    const int orow  = R + ((lane >> 4) << 3);
    const int ocol0 = Cb + (lane & 15);

    if (R + 16 <= N && Cb + 32 <= C) {
        // Fast path (wave-uniform): full tile, unconditional coalesced stores.
        const int idx0 = orow * C + ocol0;        // fits 32-bit (N*C <= 6.4M)
#pragma unroll
        for (int r = 0; r < 8; ++r) {
            Y[idx0 + r * C]      = acc0[r];
            Y[idx0 + r * C + 16] = acc1[r];
        }
    } else {
        // Ragged edge tiles only.
        const int ocol1 = ocol0 + 16;
#pragma unroll
        for (int r = 0; r < 8; ++r) {
            if (orow + r < N) {
                const int rowBase = (orow + r) * C;
                if (ocol0 < C) Y[rowBase + ocol0] = acc0[r];
                if (ocol1 < C) Y[rowBase + ocol1] = acc1[r];
            }
        }
    }
}

// ---------------------------------------------------------------------------
// el[n,h] = sum_f H[n,h,f]*al[h,f] ; er likewise. One thread per (n,h).
// ---------------------------------------------------------------------------
__global__ void attn_scores(const float* __restrict__ Hf,
                            const float* __restrict__ al,
                            const float* __restrict__ ar,
                            float* __restrict__ el, float* __restrict__ er,
                            int N, int Hh, int F) {
    const int idx = blockIdx.x * blockDim.x + threadIdx.x;
    if (idx >= N * Hh) return;
    const int n = idx / Hh, h = idx % Hh;
    const float* hp  = Hf + ((size_t)n * Hh + h) * F;
    const float* alp = al + (size_t)h * F;
    const float* arp = ar + (size_t)h * F;
    float sl = 0.f, sr = 0.f;
#pragma unroll 8
    for (int f = 0; f < F; ++f) {
        const float v = hp[f];
        sl += v * alp[f];
        sr += v * arp[f];
    }
    el[idx] = sl;
    er[idx] = sr;
}

// ---------------------------------------------------------------------------
// Per-destination-node fused leaky-relu + segment softmax + weighted gather.
// One block per dst node; thread t owns output feature (h = t/F, f = t%F).
// ---------------------------------------------------------------------------
__global__ void gat_aggregate(const float* __restrict__ Hf,
                              const float* __restrict__ el,
                              const float* __restrict__ er,
                              const int* __restrict__ row_ptr,
                              const int* __restrict__ col_idx,
                              float* __restrict__ out,
                              int N, int Hh, int F) {
    const int n = blockIdx.x;
    const int start = row_ptr[n];
    int deg = row_ptr[n + 1] - start;
    if (deg > MAXDEG) deg = MAXDEG;

    __shared__ float s_a[MAXH * MAXDEG];
    __shared__ float s_z[MAXH];
    __shared__ int   s_src[MAXDEG];

    const int t = threadIdx.x;
    for (int e = t; e < deg; e += blockDim.x) s_src[e] = col_idx[start + e];
    __syncthreads();

    if (t < deg * Hh) {                            // raw logits + leaky-relu
        const int h = t / deg;
        const int e = t % deg;
        float v = el[n * Hh + h] + er[s_src[e] * Hh + h];
        v = (v > 0.f) ? v : v * NEG_SLOPE;
        s_a[h * MAXDEG + e] = v;
    }
    __syncthreads();

    if (t < Hh) {                                  // segment softmax per head
        float m = -1e30f;
        for (int e = 0; e < deg; ++e) m = fmaxf(m, s_a[t * MAXDEG + e]);
        if (deg == 0) m = 0.f;
        float z = 0.f;
        for (int e = 0; e < deg; ++e) {
            const float a = __expf(s_a[t * MAXDEG + e] - m);
            s_a[t * MAXDEG + e] = a;
            z += a;
        }
        s_z[t] = fmaxf(z, 1e-16f);
    }
    __syncthreads();

    const int h = t / F;
    const int f = t % F;
    if (h < Hh) {
        const float invz = 1.0f / s_z[h];
        float acc = 0.f;
        for (int e = 0; e < deg; ++e) {
            // consecutive lanes read consecutive floats of one src row (L2-resident)
            acc += s_a[h * MAXDEG + e] * Hf[((size_t)s_src[e] * Hh + h) * F + f];
        }
        out[((size_t)n * Hh + h) * F + f] = acc * invz;
    }
}

// ---------------------------------------------------------------------------
extern "C" void kernel_launch(void* const* d_in, const int* in_sizes, int n_in,
                              void* d_out, int out_size, void* d_ws, size_t ws_size,
                              hipStream_t stream) {
    const float* x       = (const float*)d_in[0];
    const int*   row_ptr = (const int*)d_in[1];
    const int*   col_idx = (const int*)d_in[2];
    // d_in[3..5]: col_ptr / row_idx / permute — unused (math-identical graph)
    const float* W0  = (const float*)d_in[6];
    const float* al0 = (const float*)d_in[7];
    const float* ar0 = (const float*)d_in[8];
    const float* W1  = (const float*)d_in[9];
    const float* al1 = (const float*)d_in[10];
    const float* ar1 = (const float*)d_in[11];
    const float* W2  = (const float*)d_in[12];
    const float* al2 = (const float*)d_in[13];
    const float* ar2 = (const float*)d_in[14];
    float* out = (float*)d_out;

    const int N    = in_sizes[1] - 1;   // row_ptr has N+1 entries
    const int HID  = 64;
    const int NCLS = 40;

    float*    bufA = (float*)d_ws;                        // N*128  projected feats
    float*    bufB = bufA + (size_t)N * 128;              // N*128  aggregated feats
    float*    el   = bufB + (size_t)N * 128;              // N*2
    float*    er   = el   + (size_t)N * 2;                // N*2
    _Float16* Wt   = (_Float16*)(er + (size_t)N * 2);     // up to 128*128 halfs

    auto run_layer = [&](const float* X, const float* W, const float* al,
                         const float* ar, float* Hbuf, float* Obuf,
                         int K, int Hh, int F) {
        const int C   = Hh * F;
        const int C32 = ((C + 31) / 32) * 32;

        convert_weights<<<(C32 * K + 255) / 256, 256, 0, stream>>>(W, Wt, K, C, C32);

        const long tiles = (long)((N + 15) / 16) * (C32 / 32);
        const int wavesPerBlock = 4;
        const int nBlocks = (int)((tiles + wavesPerBlock - 1) / wavesPerBlock);
        gemm_wmma_f16<<<nBlocks, 32 * wavesPerBlock, 0, stream>>>(X, Wt, Hbuf, N, K, C, C32);

        const int nh = N * Hh;
        attn_scores<<<(nh + 255) / 256, 256, 0, stream>>>(Hbuf, al, ar, el, er, N, Hh, F);

        int bt = ((C + 31) / 32) * 32;
        if (bt < 64) bt = 64;               // cover deg*Hh score threads too
        gat_aggregate<<<N, bt, 0, stream>>>(Hbuf, el, er, row_ptr, col_idx, Obuf, N, Hh, F);
    };

    // Layer 0: x[N,128] -> bufA (proj) -> bufB (agg, N x 128)
    run_layer(x,    W0, al0, ar0, bufA, bufB, 128, 2, HID);
    // Layer 1: bufB -> bufA (proj) -> bufB (agg, N x 128)
    run_layer(bufB, W1, al1, ar1, bufA, bufB, 128, 2, HID);
    // Layer 2: bufB -> bufA (proj, N x 40) -> out (agg); mean over 1 head = identity
    run_layer(bufB, W2, al2, ar2, bufA, out, 128, 1, NCLS);
}
